// ViL_layer_61856118997523
// MI455X (gfx1250) — matrix operand order repaired and checked by reading, not compiled, not run on hardware
//
#include <hip/hip_runtime.h>

#define DIMD 384
#define HEADS 6
#define DH 64
#define KCONV 4
#define SEQ 2048
#define BATCH 8
#define NROWS (BATCH * SEQ)
#define LDH 72  // padded LDS row stride (halfs): 144B rows, 16B-aligned, conflict-free

typedef __attribute__((ext_vector_type(16))) _Float16 v16h;
typedef __attribute__((ext_vector_type(8)))  float    v8f;
typedef _Float16 h8v __attribute__((ext_vector_type(8)));

__device__ __forceinline__ v8f vzero8() {
    v8f r;
#pragma unroll
    for (int i = 0; i < 8; ++i) r[i] = 0.0f;
    return r;
}

// WMMA A-fragment: 16x32 f16 tile, rows contiguous with stride ld.
// (ISA 7.12.2) lane L: M=L%16, half=L>>4; VGPR0-3: K=half*8+0..7, VGPR4-7: K=16+half*8+0..7.
__device__ __forceinline__ v16h frag_a16(const _Float16* p0, int ld, int lane) {
    int m  = lane & 15;
    int hh = lane >> 4;
    const _Float16* p = p0 + m * ld + hh * 8;
    h8v lo = *(const h8v*)(p);
    h8v hi = *(const h8v*)(p + 16);
    v16h r;
#pragma unroll
    for (int i = 0; i < 8; ++i) { r[i] = lo[i]; r[i + 8] = hi[i]; }
    return r;
}

// WMMA B-fragment: B is 32x16 (KxN) with B[k,n] = mat[(n0+n)*ld + k0 + k].
// lane L: N=L%16, half=L>>4; VGPR0-3: K=half*8+0..7, VGPR4-7: K=16+half*8+0..7.
__device__ __forceinline__ v16h frag_b16(const _Float16* mat, int ld, int n0, int k0, int lane) {
    int n  = lane & 15;
    int kh = (lane >> 4) * 8;
    const _Float16* p = mat + (n0 + n) * ld + k0 + kh;
    h8v lo = *(const h8v*)(p);
    h8v hi = *(const h8v*)(p + 16);
    v16h r;
#pragma unroll
    for (int i = 0; i < 8; ++i) { r[i] = lo[i]; r[i + 8] = hi[i]; }
    return r;
}

__device__ __forceinline__ v8f wmma_f16(v16h a, v16h b, v8f c) {
    return __builtin_amdgcn_wmma_f32_16x16x32_f16(false, a, false, b, (short)0, c, false, false);
}

// ---------------------------------------------------------------- utilities
__global__ void f2h_kernel(const float* __restrict__ s, _Float16* __restrict__ d, int n) {
    int i = blockIdx.x * blockDim.x + threadIdx.x;
    if (i < n) d[i] = (_Float16)s[i];
}

// ------------------------------------------------------------- 1. LayerNorm
__global__ __launch_bounds__(128)
void layernorm_kernel(const float* __restrict__ x, const float* __restrict__ g,
                      const float* __restrict__ be, float* __restrict__ outf,
                      _Float16* __restrict__ outh) {
    int row = blockIdx.x;
    int t = threadIdx.x;
    __shared__ float red[128];
    float v[3];
    float s = 0.f;
#pragma unroll
    for (int i = 0; i < 3; ++i) { v[i] = x[row * DIMD + t + i * 128]; s += v[i]; }
    red[t] = s; __syncthreads();
    for (int off = 64; off > 0; off >>= 1) { if (t < off) red[t] += red[t + off]; __syncthreads(); }
    float mu = red[0] / (float)DIMD; __syncthreads();
    float s2 = 0.f;
#pragma unroll
    for (int i = 0; i < 3; ++i) { float c = v[i] - mu; s2 += c * c; }
    red[t] = s2; __syncthreads();
    for (int off = 64; off > 0; off >>= 1) { if (t < off) red[t] += red[t + off]; __syncthreads(); }
    float var = red[0] / (float)DIMD;
    float r = rsqrtf(var + (float)DIMD);   // LN_EPS == DIM
#pragma unroll
    for (int i = 0; i < 3; ++i) {
        int col = t + i * 128;
        float o = (v[i] - mu) * r * g[col] + be[col];
        outf[row * DIMD + col] = o;
        outh[row * DIMD + col] = (_Float16)o;
    }
}

// ------------------------------------- 2. pipelined WMMA GEMM  Y = X@W^T + b
// Tile: 128 (M, 8 waves x 16) x 64 (N, 4 subtiles).  Next K-step's A + 4 B
// fragments are prefetched while the current 4 WMMAs issue, so s_wait_loadcnt
// amortizes over 4 matrix ops instead of gating each one.
__global__ __launch_bounds__(256)
void gemm384_kernel(const _Float16* __restrict__ X, const _Float16* __restrict__ W,
                    const float* __restrict__ bias, float scale,
                    const float* __restrict__ resid, float* __restrict__ Yf,
                    _Float16* __restrict__ Yh, const int* __restrict__ flip_ptr,
                    int do_flip_out) {
    int wave = threadIdx.x >> 5;
    int lane = threadIdx.x & 31;
    int row0 = blockIdx.x * 128 + wave * 16;
    int n0t  = blockIdx.y * 64;
    const _Float16* Xr = X + row0 * DIMD;
    v8f acc[4];
#pragma unroll
    for (int j = 0; j < 4; ++j) acc[j] = vzero8();
    v16h a = frag_a16(Xr, DIMD, lane);
    v16h b[4];
#pragma unroll
    for (int j = 0; j < 4; ++j) b[j] = frag_b16(W, DIMD, n0t + j * 16, 0, lane);
#pragma unroll
    for (int k0 = 0; k0 < DIMD; k0 += 32) {
        v16h an = a;
        v16h bn[4] = {b[0], b[1], b[2], b[3]};
        if (k0 + 32 < DIMD) {
            an = frag_a16(Xr + k0 + 32, DIMD, lane);
#pragma unroll
            for (int j = 0; j < 4; ++j) bn[j] = frag_b16(W, DIMD, n0t + j * 16, k0 + 32, lane);
        }
#pragma unroll
        for (int j = 0; j < 4; ++j) acc[j] = wmma_f16(a, b[j], acc[j]);
        a = an;
#pragma unroll
        for (int j = 0; j < 4; ++j) b[j] = bn[j];
    }
    int flip = do_flip_out ? flip_ptr[0] : 0;
    int nloc = lane & 15, mbase = (lane >> 4) * 8;
#pragma unroll
    for (int j = 0; j < 4; ++j) {
#pragma unroll
        for (int r = 0; r < 8; ++r) {
            int row = row0 + mbase + r;
            int col = n0t + j * 16 + nloc;
            float v = (acc[j][r] + bias[col]) * scale;
            int orow = row;
            if (flip) { int bb = row / SEQ, ss = row % SEQ; orow = bb * SEQ + (SEQ - 1 - ss); }
            if (resid) v += resid[orow * DIMD + col];
            if (Yf) Yf[orow * DIMD + col] = v;
            if (Yh) Yh[orow * DIMD + col] = (_Float16)v;
        }
    }
}

// ------------------------------------------- 3. depthwise causal conv + SiLU
__global__ __launch_bounds__(256)
void conv_silu_kernel(const _Float16* __restrict__ left, const float* __restrict__ cw,
                      const float* __restrict__ cb, _Float16* __restrict__ xc) {
    int idx = blockIdx.x * blockDim.x + threadIdx.x;
    if (idx >= NROWS * DIMD) return;
    int d = idx % DIMD;
    int row = idx / DIMD;
    int s = row % SEQ;
    float acc = cb[d];
#pragma unroll
    for (int j = 0; j < KCONV; ++j) {
        int sp = s - (KCONV - 1) + j;
        if (sp >= 0) acc += (float)left[(row - s + sp) * DIMD + d] * cw[d * KCONV + j];
    }
    float sv = acc / (1.f + __expf(-acc));
    xc[idx] = (_Float16)sv;
}

// --------------------------------------------------- 4. i/f gate projections
__global__ __launch_bounds__(256)
void gates_kernel(const _Float16* __restrict__ left, const float* __restrict__ Wi,
                  const float* __restrict__ bi, const float* __restrict__ Wf,
                  const float* __restrict__ bf, float* __restrict__ ig,
                  float* __restrict__ fg) {
    int idx = blockIdx.x * blockDim.x + threadIdx.x;
    if (idx >= NROWS * HEADS) return;
    int row = idx / HEADS, h = idx % HEADS;
    float si = 0.f, sf = 0.f;
    for (int k = 0; k < DIMD; ++k) {
        float xv = (float)left[row * DIMD + k];
        si += xv * Wi[h * DIMD + k];
        sf += xv * Wf[h * DIMD + k];
    }
    si += bi[h]; sf += bf[h];
    float ls = (sf >= 0.f) ? -log1pf(__expf(-sf)) : (sf - log1pf(__expf(sf)));
    ig[row * HEADS + h] = si;
    fg[row * HEADS + h] = ls;
}

// ----------------------------------------- 5. chunkwise mLSTM scan (T = 64)
// One workgroup per (b,h); 32 sequential chunks; per chunk: 4x 64x64x64 WMMA
// GEMMs (QK^T, S@V, Q@C^T, V^T@Kw).  Q/K staged via CDNA5 async global->LDS
// DMA (ASYNCcnt); V goes through VGPRs because it is transposed on the way in.
__global__ __launch_bounds__(256)
void mlstm_scan_kernel(const _Float16* __restrict__ qg, const _Float16* __restrict__ kg,
                       const _Float16* __restrict__ vg, const float* __restrict__ ig,
                       const float* __restrict__ fg, float* __restrict__ hout) {
    __shared__ _Float16 Qs[DH * LDH], Ks[DH * LDH], Vt[DH * LDH];
    __shared__ _Float16 Cs[DH * LDH], Sw[DH * LDH], Kwt[DH * LDH];
    __shared__ float At[DH], Bsv[DH], Dec[DH], Rsum[DH], Nq[DH], Nst[DH], Bc[DH], Igc[DH];
    __shared__ float Mst;

    int bh = blockIdx.x;
    int b = bh / HEADS, h = bh % HEADS;
    int tid = threadIdx.x, wave = tid >> 5, lane = tid & 31;

    for (int i = tid; i < DH * LDH; i += 256) Cs[i] = (_Float16)0.f;
    if (tid < DH) Nst[tid] = 0.f;
    if (tid == 0) Mst = -1e30f;
    __syncthreads();

    for (int c = 0; c < SEQ / DH; ++c) {
        int s0 = c * DH;
        // stage Q,K rows via async DMA to LDS; V transposed via VGPRs
        for (int i = tid; i < DH * 8; i += 256) {        // 8 x 16B groups per row
            int t = i >> 3, j0 = (i & 7) * 8;
            int gi = (b * SEQ + s0 + t) * DIMD + h * DH + j0;
            unsigned lq = (unsigned)(size_t)(Qs + t * LDH + j0);
            unsigned lk = (unsigned)(size_t)(Ks + t * LDH + j0);
            unsigned long long aq = (unsigned long long)(size_t)(qg + gi);
            unsigned long long ak = (unsigned long long)(size_t)(kg + gi);
            asm volatile("global_load_async_to_lds_b128 %0, %1, off"
                         :: "v"(lq), "v"(aq) : "memory");
            asm volatile("global_load_async_to_lds_b128 %0, %1, off"
                         :: "v"(lk), "v"(ak) : "memory");
            h8v vv = *(const h8v*)(vg + gi);
#pragma unroll
            for (int e = 0; e < 8; ++e) Vt[(j0 + e) * LDH + t] = vv[e];
        }
        if (tid < DH) {
            Bc[tid]  = fg[(b * SEQ + s0 + tid) * HEADS + h];
            Igc[tid] = ig[(b * SEQ + s0 + tid) * HEADS + h];
        }
        asm volatile("s_wait_asynccnt 0x0" ::: "memory");
        __syncthreads();
        // serial gate prefix (64 steps, negligible): m_t = b_t + max(m0, g_t)
        if (tid == 0) {
            float run = 0.f, gmax = -1e30f, m0 = Mst;
            for (int t = 0; t < DH; ++t) {
                run += Bc[t];
                float bs = Igc[t] - run;
                if (bs > gmax) gmax = bs;
                float mt = run + fmaxf(m0, gmax);
                At[t]  = run - mt;
                Bsv[t] = bs;
                Dec[t] = __expf(run + m0 - mt);
                if (t == DH - 1) Mst = mt;
            }
        }
        __syncthreads();
        // GEMM1: scores = QK^T, masked & weighted -> Sw (f16)
        for (int tl = wave; tl < 16; tl += 8) {
            int tm = tl >> 2, tn = tl & 3;
            v8f acc = vzero8();
#pragma unroll
            for (int k0 = 0; k0 < DH; k0 += 32) {
                v16h a = frag_a16(Qs + tm * 16 * LDH + k0, LDH, lane);
                v16h bb = frag_b16(Ks, LDH, tn * 16, k0, lane);
                acc = wmma_f16(a, bb, acc);
            }
            int nloc = lane & 15, mb = (lane >> 4) * 8;
#pragma unroll
            for (int r = 0; r < 8; ++r) {
                int t = tm * 16 + mb + r, s = tn * 16 + nloc;
                float w = (s <= t) ? __expf(At[t] + Bsv[s]) : 0.f;
                Sw[t * LDH + s] = (_Float16)(acc[r] * w);
            }
        }
        __syncthreads();
        // row sums (denominator) and n0.q ; build Kwt = w_end(s)*K[s][:] transposed
        if (tid < DH) {
            float rs = 0.f;
            for (int s = 0; s < DH; ++s) rs += (float)Sw[tid * LDH + s];
            Rsum[tid] = rs;
            float nq = 0.f;
            for (int j = 0; j < DH; ++j) nq += Nst[j] * (float)Qs[tid * LDH + j];
            Nq[tid] = nq;
        }
        for (int i = tid; i < DH * DH; i += 256) {
            int s = i >> 6, j = i & 63;
            float w = __expf(At[DH - 1] + Bsv[s]);
            Kwt[j * LDH + s] = (_Float16)(w * (float)Ks[s * LDH + j]);
        }
        __syncthreads();
        // GEMM2 (intra = Sw@V) + GEMM3 (inter = Q@C^T), fused epilogue -> h
        for (int tl = wave; tl < 16; tl += 8) {
            int tm = tl >> 2, tn = tl & 3;
            v8f ai = vzero8(), an = vzero8();
#pragma unroll
            for (int k0 = 0; k0 < DH; k0 += 32) {
                v16h aS = frag_a16(Sw + tm * 16 * LDH + k0, LDH, lane);
                v16h bV = frag_b16(Vt, LDH, tn * 16, k0, lane);
                ai = wmma_f16(aS, bV, ai);
                v16h aQ = frag_a16(Qs + tm * 16 * LDH + k0, LDH, lane);
                v16h bC = frag_b16(Cs, LDH, tn * 16, k0, lane);
                an = wmma_f16(aQ, bC, an);
            }
            int nloc = lane & 15, mb = (lane >> 4) * 8;
#pragma unroll
            for (int r = 0; r < 8; ++r) {
                int t = tm * 16 + mb + r, dv = tn * 16 + nloc;
                float num = ai[r] + Dec[t] * an[r];
                float den = fmaxf(fabsf(Rsum[t] + Dec[t] * Nq[t]), 1.f);
                hout[(b * SEQ + s0 + t) * DIMD + h * DH + dv] = num / den;
            }
        }
        __syncthreads();
        // GEMM4: C = dec*C + V^T @ Kw ; n = dec*n + colsum(Kw)
        float decS = Dec[DH - 1];
        for (int tl = wave; tl < 16; tl += 8) {
            int tm = tl >> 2, tn = tl & 3;
            v8f acc = vzero8();
#pragma unroll
            for (int k0 = 0; k0 < DH; k0 += 32) {
                v16h aV = frag_a16(Vt + tm * 16 * LDH + k0, LDH, lane);
                v16h bK = frag_b16(Kwt, LDH, tn * 16, k0, lane);
                acc = wmma_f16(aV, bK, acc);
            }
            int nloc = lane & 15, mb = (lane >> 4) * 8;
#pragma unroll
            for (int r = 0; r < 8; ++r) {
                int dv = tm * 16 + mb + r, dk = tn * 16 + nloc;
                float cn = decS * (float)Cs[dv * LDH + dk] + acc[r];
                Cs[dv * LDH + dk] = (_Float16)cn;
            }
        }
        if (tid < DH) {
            float s = 0.f;
            for (int r = 0; r < DH; ++r) s += (float)Kwt[tid * LDH + r];
            Nst[tid] = decS * Nst[tid] + s;
        }
        __syncthreads();
    }
}

// --------------------------- 6. per-head GroupNorm + unflip + multiply by out
__global__ __launch_bounds__(64)
void groupnorm_fuse_kernel(const float* __restrict__ hin, const float* __restrict__ gn_g,
                           const float* __restrict__ gn_b, const float* __restrict__ outf,
                           _Float16* __restrict__ tmp_h, const int* __restrict__ flip_ptr) {
    int blk = blockIdx.x;                 // b*SEQ*H + s*H + h
    int h = blk % HEADS;
    int row = blk / HEADS;
    int b = row / SEQ, s = row % SEQ;
    int j = threadIdx.x;
    __shared__ float red[64];
    float v = hin[row * DIMD + h * DH + j];
    red[j] = v; __syncthreads();
    for (int off = 32; off > 0; off >>= 1) { if (j < off) red[j] += red[j + off]; __syncthreads(); }
    float mu = red[0] / (float)DH; __syncthreads();
    float xc = v - mu;
    red[j] = xc * xc; __syncthreads();
    for (int off = 32; off > 0; off >>= 1) { if (j < off) red[j] += red[j + off]; __syncthreads(); }
    float var = red[0] / (float)DH;
    float hn = xc * rsqrtf(var + 1e-5f) * gn_g[h * DH + j] + gn_b[h * DH + j];
    int flip = flip_ptr[0];
    int so = flip ? (SEQ - 1 - s) : s;
    int oidx = (b * SEQ + so) * DIMD + h * DH + j;
    tmp_h[oidx] = (_Float16)(hn * outf[oidx]);
}

// ============================================================== launcher
extern "C" void kernel_launch(void* const* d_in, const int* in_sizes, int n_in,
                              void* d_out, int out_size, void* d_ws, size_t ws_size,
                              hipStream_t stream) {
    const float* x      = (const float*)d_in[0];
    const float* ln_g   = (const float*)d_in[1];
    const float* ln_b   = (const float*)d_in[2];
    const float* W2     = (const float*)d_in[5];
    const float* b2     = (const float*)d_in[6];
    const float* W3     = (const float*)d_in[7];
    const float* b3     = (const float*)d_in[8];
    const float* conv_w = (const float*)d_in[9];
    const float* conv_b = (const float*)d_in[10];
    const float* Wq     = (const float*)d_in[11];
    const float* bq     = (const float*)d_in[12];
    const float* Wk     = (const float*)d_in[13];
    const float* bk     = (const float*)d_in[14];
    const float* Wv     = (const float*)d_in[15];
    const float* bv     = (const float*)d_in[16];
    const float* Wi     = (const float*)d_in[17];
    const float* bi     = (const float*)d_in[18];
    const float* Wf     = (const float*)d_in[19];
    const float* bf     = (const float*)d_in[20];
    const float* gn_g   = (const float*)d_in[21];
    const float* gn_b   = (const float*)d_in[22];
    const int*   flip   = (const int*)d_in[23];
    float* out = (float*)d_out;

    // workspace layout
    char* ws = (char*)d_ws;
    size_t off = 0;
    auto nxt = [&](size_t bytes) { size_t r = off; off = (off + bytes + 255) & ~(size_t)255; return r; };
    const size_t WB = (size_t)DIMD * DIMD * 2;      // one f16 weight matrix
    const size_t AH = (size_t)NROWS * DIMD * 2;     // f16 activation plane
    const size_t AF = (size_t)NROWS * DIMD * 4;     // f32 activation plane
    _Float16* W2h   = (_Float16*)(ws + nxt(WB));
    _Float16* W3h   = (_Float16*)(ws + nxt(WB));
    _Float16* Wqh   = (_Float16*)(ws + nxt(WB));
    _Float16* Wkh   = (_Float16*)(ws + nxt(WB));
    _Float16* Wvh   = (_Float16*)(ws + nxt(WB));
    float*    outf  = (float*)   (ws + nxt(AF));
    _Float16* outh  = (_Float16*)(ws + nxt(AH));    // reused as tmp_h later
    _Float16* lefth = (_Float16*)(ws + nxt(AH));
    _Float16* xch   = (_Float16*)(ws + nxt(AH));
    _Float16* qh    = (_Float16*)(ws + nxt(AH));
    _Float16* kh    = (_Float16*)(ws + nxt(AH));
    _Float16* vh    = (_Float16*)(ws + nxt(AH));
    float*    igb   = (float*)   (ws + nxt((size_t)NROWS * HEADS * 4));
    float*    fgb   = (float*)   (ws + nxt((size_t)NROWS * HEADS * 4));
    float*    hf    = (float*)   (ws + nxt(AF));
    _Float16* tmph  = outh;  // reuse: outh only needed for the W2 GEMM

    const int WN = DIMD * DIMD;
    f2h_kernel<<<(WN + 255) / 256, 256, 0, stream>>>(W2, W2h, WN);
    f2h_kernel<<<(WN + 255) / 256, 256, 0, stream>>>(W3, W3h, WN);
    f2h_kernel<<<(WN + 255) / 256, 256, 0, stream>>>(Wq, Wqh, WN);
    f2h_kernel<<<(WN + 255) / 256, 256, 0, stream>>>(Wk, Wkh, WN);
    f2h_kernel<<<(WN + 255) / 256, 256, 0, stream>>>(Wv, Wvh, WN);

    layernorm_kernel<<<NROWS, 128, 0, stream>>>(x, ln_g, ln_b, outf, outh);

    dim3 ggrid(NROWS / 128, DIMD / 64);
    gemm384_kernel<<<ggrid, 256, 0, stream>>>(outh, W2h, b2, 1.f, nullptr, nullptr, lefth, flip, 1);
    conv_silu_kernel<<<(NROWS * DIMD + 255) / 256, 256, 0, stream>>>(lefth, conv_w, conv_b, xch);
    gates_kernel<<<(NROWS * HEADS + 255) / 256, 256, 0, stream>>>(lefth, Wi, bi, Wf, bf, igb, fgb);
    gemm384_kernel<<<ggrid, 256, 0, stream>>>(xch,   Wqh, bq, 1.f,    nullptr, nullptr, qh, flip, 0);
    gemm384_kernel<<<ggrid, 256, 0, stream>>>(xch,   Wkh, bk, 0.125f, nullptr, nullptr, kh, flip, 0);
    gemm384_kernel<<<ggrid, 256, 0, stream>>>(lefth, Wvh, bv, 1.f,    nullptr, nullptr, vh, flip, 0);
    mlstm_scan_kernel<<<BATCH * HEADS, 256, 0, stream>>>(qh, kh, vh, igb, fgb, hf);
    groupnorm_fuse_kernel<<<NROWS * HEADS, 64, 0, stream>>>(hf, gn_g, gn_b, outf, tmph, flip);
    gemm384_kernel<<<ggrid, 256, 0, stream>>>(tmph, W3h, b3, 1.f, x, out, nullptr, flip, 0);
}
